// ALiBiAttention_5162550690186
// MI455X (gfx1250) — matrix operand order repaired and checked by reading, not compiled, not run on hardware
//
#include <hip/hip_runtime.h>
#include <math.h>

#define HQ   16
#define HKV  4
#define HD   64
#define DIMK 1024
#define WIN  16

typedef __attribute__((ext_vector_type(16))) __bf16 v16bf;
typedef __attribute__((ext_vector_type(8)))  __bf16 v8bf;
typedef __attribute__((ext_vector_type(8)))  float  v8f;

__device__ __forceinline__ unsigned short f2bf(float f) {
    unsigned int u = __builtin_bit_cast(unsigned int, f);
    u += 0x7FFFu + ((u >> 16) & 1u);           // round-to-nearest-even
    return (unsigned short)(u >> 16);
}

// Build a 16xK A operand register from two 8-element (16B) chunks
__device__ __forceinline__ v16bf a_combine(v8bf lo, v8bf hi) {
    v16bf r;
#pragma unroll
    for (int i = 0; i < 8; ++i) { r[i] = lo[i]; r[i + 8] = hi[i]; }
    return r;
}

// A-operand load: lane&15 = row; lane>>4 selects K chunks {0..15} / {16..31}
__device__ __forceinline__ v16bf load_a(const unsigned short* rowp, int lane) {
    int koff = (lane >> 4) << 4;  // 0 or 16
    v8bf lo = *(const v8bf*)(rowp + koff);
    v8bf hi = *(const v8bf*)(rowp + koff + 8);
    return a_combine(lo, hi);
}

// ---------------- fp32 -> bf16 convert ----------------
__global__ void cvt_bf16_kernel(const float* __restrict__ in, unsigned short* __restrict__ out, int n) {
    int i = blockIdx.x * blockDim.x + threadIdx.x;
    if (i < n) out[i] = f2bf(in[i]);
}

// ---------------- WMMA GEMM: C[M,N] = A[M,K](bf16) * W[N,K]^T(bf16), fp32 out ----------
// one wave computes a 32x64 output block: each B tile is reused by 2 A tiles
// -> 8 WMMA per K-step vs 12 b128 loads (1.7x higher flop/byte than 16x64 tiling)
__global__ void gemm_bf16_kernel(const unsigned short* __restrict__ A,
                                 const unsigned short* __restrict__ W,
                                 float* __restrict__ C,
                                 int M, int N, int K) {
    int wave = (int)((blockIdx.x * blockDim.x + threadIdx.x) >> 5);
    int lane = threadIdx.x & 31;
    int nBlocks = N >> 6;
    int mTile = wave / nBlocks;
    int nBlk  = wave - mTile * nBlocks;
    if (mTile >= (M >> 5)) return;               // wave-uniform
    int m0 = mTile << 5, n0 = nBlk << 6;

    const unsigned short* abase0 = A + (size_t)(m0 + (lane & 15)) * K;
    const unsigned short* abase1 = abase0 + (size_t)16 * K;
    int bhalf = (lane >> 4) << 4;                // 0 or 16 (K offset for B)

    v8f acc[2][4] = {};
    for (int k0 = 0; k0 < K; k0 += 32) {
        // prefetch next K-block of the two A streams (global_prefetch_b8)
        if (k0 + 128 < K) {
            __builtin_prefetch((const void*)(abase0 + k0 + 128), 0, 1);
            __builtin_prefetch((const void*)(abase1 + k0 + 128), 0, 1);
        }
        v16bf a0 = load_a(abase0 + k0, lane);
        v16bf a1 = load_a(abase1 + k0, lane);
#pragma unroll
        for (int t = 0; t < 4; ++t) {
            int n = n0 + (t << 4) + (lane & 15);
            v16bf bv = *(const v16bf*)(W + (size_t)n * K + k0 + bhalf);
            acc[0][t] = __builtin_amdgcn_wmma_f32_16x16x32_bf16(
                false, a0, false, bv, (short)0, acc[0][t], false, false);
            acc[1][t] = __builtin_amdgcn_wmma_f32_16x16x32_bf16(
                false, a1, false, bv, (short)0, acc[1][t], false, false);
        }
    }
    int cb = lane & 15;
#pragma unroll
    for (int u = 0; u < 2; ++u) {
        int rbase = m0 + (u << 4) + ((lane >> 4) << 3);
#pragma unroll
        for (int t = 0; t < 4; ++t)
#pragma unroll
            for (int r = 0; r < 8; ++r)
                C[(size_t)(rbase + r) * N + n0 + (t << 4) + cb] = acc[u][t][r];
    }
}

// ---------------- per-head RMSNorm (64-dim vectors), fp32 in -> bf16 out -------------
__global__ void rmsnorm_bf16_kernel(const float* __restrict__ x, const float* __restrict__ w,
                                    unsigned short* __restrict__ y, int nvec) {
    int wave = (int)((blockIdx.x * blockDim.x + threadIdx.x) >> 5);
    int lane = threadIdx.x & 31;
    if (wave >= nvec) return;
    float2 v = ((const float2*)(x + (size_t)wave * 64))[lane];
    float ss = v.x * v.x + v.y * v.y;
#pragma unroll
    for (int m = 16; m >= 1; m >>= 1) ss += __shfl_xor(ss, m, 32);
    float sc = rsqrtf(ss * (1.0f / 64.0f) + 0.01f);
    float2 wv = ((const float2*)w)[lane];
    y[(size_t)wave * 64 + lane * 2]     = f2bf(v.x * sc * wv.x);
    y[(size_t)wave * 64 + lane * 2 + 1] = f2bf(v.y * sc * wv.y);
}

// ---------------- V transpose: [b,l,h,d] fp32 -> [b,h,d,l] bf16 ----------------------
__global__ void vtrans_kernel(const float* __restrict__ vf, unsigned short* __restrict__ vt,
                              int B, int L) {
    int i = blockIdx.x * blockDim.x + threadIdx.x;
    int total = B * L * HKV * HD;
    if (i >= total) return;
    int d = i & 63;
    int h = (i >> 6) & 3;
    int rest = i >> 8;
    int l = rest % L;
    int b = rest / L;
    vt[(((size_t)b * HKV + h) * HD + d) * L + l] = f2bf(vf[i]);
}

// ---------------- WMMA windowed ALiBi attention -------------------------------------
// one wave: (b, head h, 16-query tile). Keys window = [q0-16, q0+15] -> 16x32 scores.
__global__ void attn_kernel(const unsigned short* __restrict__ qb,  // [b,l,hq,64] bf16
                            const unsigned short* __restrict__ kb,  // [b,l,hkv,64] bf16
                            const unsigned short* __restrict__ vt,  // [b,hkv,64,L] bf16
                            unsigned short* __restrict__ ab,        // [b,l,hq,64] bf16
                            int B, int L) {
    __shared__ unsigned short plds[4][16 * 32];
    int wave = (int)((blockIdx.x * blockDim.x + threadIdx.x) >> 5);
    int lane = threadIdx.x & 31;
    int wslot = threadIdx.x >> 5;
    int qtiles = L >> 4;
    int qt = wave % qtiles;
    int h  = (wave / qtiles) % HQ;
    int b  = wave / (qtiles * HQ);
    if (b >= B) return;                           // wave-uniform
    int kvh = h % HKV;                            // GQA: head h -> kv head h%4
    int q0 = qt << 4;
    int jb = q0 - WIN;                            // key window base
    float slope = exp2f(-0.5f * (float)(h + 1));  // ALiBi, 16 heads

    // Q tile A-operands, 2 chunks of head_dim (K=0..31, 32..63)
    int lq = q0 + (lane & 15);
    const unsigned short* qrow = qb + (((size_t)b * L + lq) * HQ + h) * HD;
    v16bf qa0 = load_a(qrow, lane);
    v16bf qa1 = load_a(qrow + 32, lane);

    // scores = Q * K^T : two 16x16 key sub-tiles, K-dim = 64 (2 WMMA each)
    int bhalf = (lane >> 4) << 4;
    v8f s[2] = {};
#pragma unroll
    for (int t = 0; t < 2; ++t) {
        int j = jb + (t << 4) + (lane & 15);
        int jc = j < 0 ? 0 : j;
        const unsigned short* krow = kb + (((size_t)b * L + jc) * HKV + kvh) * HD;
        v16bf kb0 = *(const v16bf*)(krow + bhalf);
        v16bf kb1 = *(const v16bf*)(krow + 32 + bhalf);
        s[t] = __builtin_amdgcn_wmma_f32_16x16x32_bf16(false, qa0, false, kb0, (short)0, s[t], false, false);
        s[t] = __builtin_amdgcn_wmma_f32_16x16x32_bf16(false, qa1, false, kb1, (short)0, s[t], false, false);
    }

    // bias + mask + softmax; D layout: VGPR r = row r (+8 in upper half), lane&15 = col
    const float scale = 0.125f;  // 1/sqrt(64)
    int col = lane & 15;
#pragma unroll
    for (int r = 0; r < 8; ++r) {
        int m = r + ((lane >> 4) << 3);
        int rel0 = col - 16 - m;                 // keys jb..jb+15  (always <= -1)
        int rel1 = col - m;                      // keys jb+16..jb+31
        int j0 = jb + col;
        bool ok0 = (j0 >= 0) && (rel0 >= -WIN);
        bool ok1 = (rel1 <= 0);
        float v0 = ok0 ? s[0][r] * scale + slope * (float)rel0 : -INFINITY;
        float v1 = ok1 ? s[1][r] * scale + slope * (float)rel1 : -INFINITY;
        float mx = fmaxf(v0, v1);
#pragma unroll
        for (int msk = 8; msk >= 1; msk >>= 1) mx = fmaxf(mx, __shfl_xor(mx, msk, 32));
        float e0 = __expf(v0 - mx);
        float e1 = __expf(v1 - mx);
        float sum = e0 + e1;
#pragma unroll
        for (int msk = 8; msk >= 1; msk >>= 1) sum += __shfl_xor(sum, msk, 32);
        float inv = __frcp_rn(sum);
        plds[wslot][m * 32 + col]      = f2bf(e0 * inv);
        plds[wslot][m * 32 + 16 + col] = f2bf(e1 * inv);
    }

    // reload probs in A layout (16 rows x 32 keys) — same-wave LDS RAW, DScnt handled
    v16bf pa = load_a(&plds[wslot][(lane & 15) * 32], lane);

    // out = P * V : 4 N-subtiles of head_dim; V^T layout makes B loads contiguous in j
    int jrow = jb + ((lane >> 4) << 4);
    if (jrow < 0) jrow = 0;                      // probs for j<0 are zero anyway
    v8f o[4];
#pragma unroll
    for (int t = 0; t < 4; ++t) {
        int d = (t << 4) + (lane & 15);
        v16bf bv = *(const v16bf*)(vt + ((((size_t)b * HKV + kvh) * HD + d) * L) + jrow);
        v8f z = {};
        o[t] = __builtin_amdgcn_wmma_f32_16x16x32_bf16(false, pa, false, bv, (short)0, z, false, false);
    }
#pragma unroll
    for (int t = 0; t < 4; ++t)
#pragma unroll
        for (int r = 0; r < 8; ++r) {
            int lrow = q0 + r + ((lane >> 4) << 3);
            ab[(((size_t)b * L + lrow) * HQ + h) * HD + (t << 4) + (lane & 15)] = f2bf(o[t][r]);
        }
}

// =====================================================================================
extern "C" void kernel_launch(void* const* d_in, const int* in_sizes, int n_in,
                              void* d_out, int out_size, void* d_ws, size_t ws_size,
                              hipStream_t stream) {
    const float* x   = (const float*)d_in[0];
    const float* wq  = (const float*)d_in[1];
    const float* wk  = (const float*)d_in[2];
    const float* wv  = (const float*)d_in[3];
    const float* wo  = (const float*)d_in[4];
    const float* qnw = (const float*)d_in[5];
    const float* knw = (const float*)d_in[6];
    float* out = (float*)d_out;

    const int B = 2, L = 2048;
    const int M = B * L;                 // 4096
    const int NKV = HKV * HD;            // 256

    char* ws = (char*)d_ws;
    size_t off = 0;
    auto take = [&](size_t bytes) { void* p = ws + off; off += (bytes + 255) & ~(size_t)255; return p; };
    unsigned short* xb  = (unsigned short*)take((size_t)M * DIMK * 2);
    unsigned short* wqb = (unsigned short*)take((size_t)DIMK * DIMK * 2);
    unsigned short* wkb = (unsigned short*)take((size_t)NKV * DIMK * 2);
    unsigned short* wvb = (unsigned short*)take((size_t)NKV * DIMK * 2);
    unsigned short* wob = (unsigned short*)take((size_t)DIMK * DIMK * 2);
    float* qf = (float*)take((size_t)M * DIMK * 4);
    float* kf = (float*)take((size_t)M * NKV * 4);
    float* vf = (float*)take((size_t)M * NKV * 4);
    unsigned short* qb2 = (unsigned short*)take((size_t)M * DIMK * 2);
    unsigned short* kb2 = (unsigned short*)take((size_t)M * NKV * 2);
    unsigned short* vtb = (unsigned short*)take((size_t)M * NKV * 2);
    unsigned short* ab  = (unsigned short*)take((size_t)M * DIMK * 2);

    // 1) fp32 -> bf16 conversions
    auto cvt = [&](const float* src, unsigned short* dst, int n) {
        cvt_bf16_kernel<<<(n + 255) / 256, 256, 0, stream>>>(src, dst, n);
    };
    cvt(x,  xb,  M * DIMK);
    cvt(wq, wqb, DIMK * DIMK);
    cvt(wk, wkb, NKV * DIMK);
    cvt(wv, wvb, NKV * DIMK);
    cvt(wo, wob, DIMK * DIMK);

    // 2) QKV projections (WMMA GEMM, 32x64 per-wave tiles)
    auto gemm = [&](const unsigned short* A, const unsigned short* W, float* C,
                    int m, int n, int k) {
        int waves = (m >> 5) * (n >> 6);
        gemm_bf16_kernel<<<(waves + 7) / 8, 256, 0, stream>>>(A, W, C, m, n, k);
    };
    gemm(xb, wqb, qf, M, DIMK, DIMK);
    gemm(xb, wkb, kf, M, NKV,  DIMK);
    gemm(xb, wvb, vf, M, NKV,  DIMK);

    // 3) per-head RMSNorm -> bf16
    {
        int nvq = M * HQ;    // 65536 vectors of 64
        int nvk = M * HKV;   // 16384
        rmsnorm_bf16_kernel<<<(nvq + 3) / 4, 128, 0, stream>>>(qf, qnw, qb2, nvq);
        rmsnorm_bf16_kernel<<<(nvk + 3) / 4, 128, 0, stream>>>(kf, knw, kb2, nvk);
    }

    // 4) V transpose -> [b,h,d,L] bf16
    {
        int n = M * NKV;
        vtrans_kernel<<<(n + 255) / 256, 256, 0, stream>>>(vf, vtb, B, L);
    }

    // 5) windowed ALiBi attention (WMMA)
    {
        int waves = B * HQ * (L >> 4);   // 4096
        attn_kernel<<<(waves + 3) / 4, 128, 0, stream>>>(qb2, kb2, vtb, ab, B, L);
    }

    // 6) output projection (WMMA GEMM) -> fp32 d_out
    gemm(ab, wob, out, M, DIMK, DIMK);
}
